// ConditionalNeuralNetwork_20590073217531
// MI455X (gfx1250) — compile-verified, matrix operand order checked by reading, not run on hardware
//
#include <hip/hip_runtime.h>

// ---------------------------------------------------------------------------
// Problem constants (match reference)
// ---------------------------------------------------------------------------
#define BN   16384
#define DIN  256
#define S1   1024
#define S2   1024
#define H1   512
#define H2   256
#define CC   4

typedef __attribute__((ext_vector_type(16))) __bf16          v16bf;
typedef __attribute__((ext_vector_type(8)))  float           v8f;
typedef __attribute__((ext_vector_type(8)))  unsigned short  ushort8;

union FragU { v16bf v; ushort8 h[2]; };

static __device__ __forceinline__ unsigned short f32_to_bf16(float f) {
    unsigned int u = __float_as_uint(f);
    unsigned int r = (u + 0x7FFFu + ((u >> 16) & 1u)) >> 16;  // RNE
    return (unsigned short)r;
}
static __device__ __forceinline__ float bf16_to_f32(unsigned short h) {
    return __uint_as_float(((unsigned int)h) << 16);
}

// Load 16 bytes from LDS at a raw byte offset (addrspace(3) inttoptr:
// integer provenance => compiler must emit a real ds_load and must assume
// it aliases the inline-asm DMA writes).
static __device__ __forceinline__ ushort8 lds_read8(unsigned byteOff) {
    typedef __attribute__((address_space(3))) const ushort8* p3_t;
    return *(p3_t)byteOff;
}

// ---------------------------------------------------------------------------
// Elementwise f32 -> bf16
// ---------------------------------------------------------------------------
__global__ void cvt_f32_to_bf16(const float* __restrict__ in,
                                unsigned short* __restrict__ out,
                                long long n) {
    long long i = (long long)blockIdx.x * blockDim.x + threadIdx.x;
    long long stride = (long long)gridDim.x * blockDim.x;
    for (; i < n; i += stride) out[i] = f32_to_bf16(in[i]);
}

// ---------------------------------------------------------------------------
// Batched transpose + convert: in [batches, K, N] f32 -> out [batches, N, K] bf16
// ---------------------------------------------------------------------------
__global__ void transpose_cvt_bf16(const float* __restrict__ in,
                                   unsigned short* __restrict__ out,
                                   int Kd, int Nd, int batches) {
    long long planar = (long long)Kd * Nd;
    long long total  = planar * batches;
    long long stride = (long long)gridDim.x * blockDim.x;
    for (long long i = (long long)blockIdx.x * blockDim.x + threadIdx.x;
         i < total; i += stride) {
        long long b   = i / planar;
        long long rem = i - b * planar;
        int k = (int)(rem / Nd);
        int n = (int)(rem - (long long)k * Nd);
        out[b * planar + (long long)n * Kd + k] = f32_to_bf16(in[i]);
    }
}

// ---------------------------------------------------------------------------
// Batched bf16 WMMA GEMM:  Out[c] = act( A[c] @ Bt[c]^T + bias[c] )
//   A  : [M, K]  bf16 row-major,  Bt : [N, K]  bf16 row-major (pre-transposed)
//   Out: [M, N]  bf16 row-major
// Block: 256 threads = 8 waves; tile 128(M) x 128(N); K step 32.
// Double-buffered LDS filled by async global->LDS DMA (ASYNCcnt), one
// barrier per K-step; WMMA overlaps the in-flight DMA of the next stage.
// ---------------------------------------------------------------------------
__global__ __launch_bounds__(256)
void gemm_bias_act_bf16(const unsigned short* __restrict__ A,  long long sA,
                        const unsigned short* __restrict__ Bt, long long sB,
                        const float* __restrict__ bias,        long long sBias,
                        unsigned short* __restrict__ Out,      long long sOut,
                        int M, int N, int K, int relu) {
    // single static LDS allocation -> base offset 0 in the group segment.
    // buffer s at bytes [s*16384, s*16384+16384): A tile 8KB then B tile 8KB.
    __shared__ __align__(16) unsigned short lds[2 * 8192];   // 32 KB

    const int tid  = threadIdx.x;
    const int lane = tid & 31;
    const int wave = tid >> 5;
    const int c    = blockIdx.z;

    A    += (long long)c * sA;
    Bt   += (long long)c * sB;
    bias += (long long)c * sBias;
    Out  += (long long)c * sOut;

    const int row0 = blockIdx.x * 128;
    const int col0 = blockIdx.y * 128;

    const int mBase = (wave & 3) * 32;   // wave's row offset in tile
    const int nBase = (wave >> 2) * 64;  // wave's col offset in tile

    v8f acc[2][4];
#pragma unroll
    for (int i = 0; i < 2; ++i)
#pragma unroll
        for (int j = 0; j < 4; ++j) acc[i][j] = (v8f)(0.0f);

    // global->LDS mapping: each thread moves 16 halves (32B) of one tile row
    const int ldRow = tid >> 1;          // 0..127
    const int ldSeg = (tid & 1) * 16;    // 0 or 16 (halves)

    // WMMA 16-bit fragment layout: lanes 0-15 take K 0-7 & 16-23,
    // lanes 16-31 take K 8-15 & 24-31 (of the 32-wide K slab)
    const int hsel = (lane >> 4) * 8;    // 0 or 8
    const int lrow = lane & 15;

    // async DMA of one 128x32 A slab + 128x32 B slab into LDS buffer s.
    // The unused pointer operand escapes `lds` so the allocation stays live
    // and the "memory" clobber is known to alias it.
    auto prefetch = [&](int k0, int s) {
        const unsigned short* ga = A  + (long long)(row0 + ldRow) * K + k0 + ldSeg;
        const unsigned short* gb = Bt + (long long)(col0 + ldRow) * K + k0 + ldSeg;
        unsigned la = (unsigned)((s * 8192 + ldRow * 32 + ldSeg) * 2);   // bytes
        unsigned lb = la + 4096u * 2u;
        asm volatile(
            "global_load_async_to_lds_b128 %0, %4, off\n\t"
            "global_load_async_to_lds_b128 %1, %5, off\n\t"
            "global_load_async_to_lds_b128 %2, %6, off\n\t"
            "global_load_async_to_lds_b128 %3, %7, off"
            :
            : "v"(la), "v"(la + 16u), "v"(lb), "v"(lb + 16u),
              "v"((unsigned long long)(uintptr_t)ga),
              "v"((unsigned long long)(uintptr_t)(ga + 8)),
              "v"((unsigned long long)(uintptr_t)gb),
              "v"((unsigned long long)(uintptr_t)(gb + 8)),
              "v"((unsigned short*)lds)           // escape LDS allocation
            : "memory");
    };

    int cur = 0;
    prefetch(0, 0);

    for (int k0 = 0; k0 < K; k0 += 32) {
        // my DMA into buf[cur] is done -> signal; barrier -> everyone's done
        asm volatile("s_wait_asynccnt 0" ::: "memory");
        __syncthreads();

        // kick off next stage into the other buffer (readers of it finished
        // before they reached this barrier)
        if (k0 + 32 < K) prefetch(k0 + 32, cur ^ 1);

        const unsigned baseA = (unsigned)(cur * 16384);      // bytes
        const unsigned baseB = baseA + 8192u;                // bytes

        FragU aF[2], bF[4];
#pragma unroll
        for (int i = 0; i < 2; ++i) {
            int r = mBase + i * 16 + lrow;
            aF[i].h[0] = lds_read8(baseA + (unsigned)((r * 32 + hsel) * 2));
            aF[i].h[1] = lds_read8(baseA + (unsigned)((r * 32 + 16 + hsel) * 2));
        }
#pragma unroll
        for (int j = 0; j < 4; ++j) {
            int r = nBase + j * 16 + lrow;
            bF[j].h[0] = lds_read8(baseB + (unsigned)((r * 32 + hsel) * 2));
            bF[j].h[1] = lds_read8(baseB + (unsigned)((r * 32 + 16 + hsel) * 2));
        }
#pragma unroll
        for (int i = 0; i < 2; ++i)
#pragma unroll
            for (int j = 0; j < 4; ++j)
                acc[i][j] = __builtin_amdgcn_wmma_f32_16x16x32_bf16(
                    false, aF[i].v, false, bF[j].v,
                    (short)0, acc[i][j], false, false);

        cur ^= 1;
    }

    // Epilogue: C layout -> VGPR r holds M=r (lanes 0-15) / M=r+8 (lanes 16-31)
    const int laneRowOfs = (lane < 16) ? 0 : 8;
#pragma unroll
    for (int j = 0; j < 4; ++j) {
        int col = col0 + nBase + j * 16 + lrow;
        float bv = bias[col];
#pragma unroll
        for (int i = 0; i < 2; ++i) {
            int rbase = row0 + mBase + i * 16 + laneRowOfs;
#pragma unroll
            for (int r = 0; r < 8; ++r) {
                float v = acc[i][j][r] + bv;
                if (relu) v = v > 0.0f ? v : 0.0f;
                Out[(long long)(rbase + r) * N + col] = f32_to_bf16(v);
            }
        }
    }
}

// ---------------------------------------------------------------------------
// Final: per-sample head select + GEMV(256) + sigmoid. One wave per sample.
//   a2 : [C, B, H2] bf16, HW3: [C, H2] f32, Hb3: [C] f32
// ---------------------------------------------------------------------------
__global__ __launch_bounds__(256)
void logit_select_sigmoid(const unsigned short* __restrict__ a2,
                          const int* __restrict__ ff,
                          const float* __restrict__ HW3,
                          const float* __restrict__ Hb3,
                          float* __restrict__ out) {
    int gwave = (int)((blockIdx.x * blockDim.x + threadIdx.x) >> 5);
    int lane  = threadIdx.x & 31;
    if (gwave >= BN) return;

    int c = ff[gwave * 2 + 0] * 2 + ff[gwave * 2 + 1];
    const unsigned short* row = a2 + ((long long)c * BN + gwave) * H2;
    const float* w = HW3 + (long long)c * H2;

    float s = 0.0f;
#pragma unroll
    for (int q = 0; q < 8; ++q) {
        int k = lane * 8 + q;                 // H2 = 256 = 32 lanes * 8
        s += bf16_to_f32(row[k]) * w[k];
    }
#pragma unroll
    for (int off = 16; off > 0; off >>= 1) s += __shfl_xor(s, off, 32);

    if (lane == 0) {
        float z = s + Hb3[c];
        out[gwave] = 1.0f / (1.0f + __expf(-z));
    }
}

// ---------------------------------------------------------------------------
// Launch
// ---------------------------------------------------------------------------
extern "C" void kernel_launch(void* const* d_in, const int* in_sizes, int n_in,
                              void* d_out, int out_size, void* d_ws, size_t ws_size,
                              hipStream_t stream) {
    (void)in_sizes; (void)n_in; (void)out_size; (void)ws_size;

    const float* x   = (const float*)d_in[0];
    const int*   ff  = (const int*)d_in[1];
    const float* W1  = (const float*)d_in[2];
    const float* b1  = (const float*)d_in[3];
    const float* W2  = (const float*)d_in[4];
    const float* b2  = (const float*)d_in[5];
    const float* HW1 = (const float*)d_in[6];
    const float* Hb1 = (const float*)d_in[7];
    const float* HW2 = (const float*)d_in[8];
    const float* Hb2 = (const float*)d_in[9];
    const float* HW3 = (const float*)d_in[10];
    const float* Hb3 = (const float*)d_in[11];
    float* out = (float*)d_out;

    // workspace carve-up (256B aligned)
    char* ws = (char*)d_ws;
    size_t off = 0;
    auto carve = [&](size_t bytes) -> char* {
        char* p = ws + off;
        off = (off + bytes + 255) & ~(size_t)255;
        return p;
    };
    unsigned short* xb   = (unsigned short*)carve((size_t)BN * DIN * 2);
    unsigned short* w1t  = (unsigned short*)carve((size_t)S1 * DIN * 2);
    unsigned short* w2t  = (unsigned short*)carve((size_t)S2 * S1 * 2);
    unsigned short* hw1t = (unsigned short*)carve((size_t)CC * H1 * S2 * 2);
    unsigned short* hw2t = (unsigned short*)carve((size_t)CC * H2 * H1 * 2);
    unsigned short* h1   = (unsigned short*)carve((size_t)BN * S1 * 2);
    unsigned short* h2   = (unsigned short*)carve((size_t)BN * S2 * 2);
    unsigned short* a1   = (unsigned short*)carve((size_t)CC * BN * H1 * 2);
    unsigned short* a2   = (unsigned short*)carve((size_t)CC * BN * H2 * 2);

    // --- conversions (bf16 path) ---
    {
        long long n = (long long)BN * DIN;
        cvt_f32_to_bf16<<<dim3(4096), dim3(256), 0, stream>>>(x, xb, n);
    }
    transpose_cvt_bf16<<<dim3(1024), dim3(256), 0, stream>>>(W1, w1t, DIN, S1, 1);
    transpose_cvt_bf16<<<dim3(2048), dim3(256), 0, stream>>>(W2, w2t, S1, S2, 1);
    transpose_cvt_bf16<<<dim3(2048), dim3(256), 0, stream>>>(HW1, hw1t, S2, H1, CC);
    transpose_cvt_bf16<<<dim3(1024), dim3(256), 0, stream>>>(HW2, hw2t, H1, H2, CC);

    // --- trunk layer 1: h1 = relu(x @ W1 + b1)  [16384,1024] ---
    gemm_bias_act_bf16<<<dim3(BN / 128, S1 / 128, 1), dim3(256), 0, stream>>>(
        xb, 0, w1t, 0, b1, 0, h1, 0, BN, S1, DIN, 1);

    // --- trunk layer 2: h2 = relu(h1 @ W2 + b2)  [16384,1024] ---
    gemm_bias_act_bf16<<<dim3(BN / 128, S2 / 128, 1), dim3(256), 0, stream>>>(
        h1, 0, w2t, 0, b2, 0, h2, 0, BN, S2, S1, 1);

    // --- head layer 1 (batched over 4 combos, A stride 0 reuses h2) ---
    gemm_bias_act_bf16<<<dim3(BN / 128, H1 / 128, CC), dim3(256), 0, stream>>>(
        h2, 0,
        hw1t, (long long)H1 * S2,
        Hb1,  (long long)H1,
        a1,   (long long)BN * H1,
        BN, H1, S2, 1);

    // --- head layer 2 (batched) ---
    gemm_bias_act_bf16<<<dim3(BN / 128, H2 / 128, CC), dim3(256), 0, stream>>>(
        a1,   (long long)BN * H1,
        hw2t, (long long)H2 * H1,
        Hb2,  (long long)H2,
        a2,   (long long)BN * H2,
        BN, H2, H1, 1);

    // --- select + final dot + sigmoid: one wave per sample ---
    logit_select_sigmoid<<<dim3((BN * 32) / 256), dim3(256), 0, stream>>>(
        a2, ff, HW3, Hb3, out);
}
